// ChirpTokenizer_35656818491859
// MI455X (gfx1250) — matrix operand order repaired and checked by reading, not compiled, not run on hardware
//
#include <hip/hip_runtime.h>
#include <hip/hip_bf16.h>

// ---------------- problem constants (match reference) ----------------
#define K_WIN   1024
#define HOP     512
#define KTAU    2048
#define FK      1025
#define NCOL    2050          // 2*FK (re,im interleaved columns)
#define D_DIM   16
#define NWIN    511           // (262144 - 1024)/512 + 1
#define NB      4
#define NSAMP   262144
#define NT_CT   129           // ceil(NCOL/16)
#define NKC     64            // KTAU / 32 (bf16 WMMA K = 32)
#define A_STRIDE 2056         // padded bf16 row stride in LDS (16B aligned, bank-conflict free)

typedef __attribute__((ext_vector_type(16))) __bf16 v16bf;
typedef __attribute__((ext_vector_type(8)))  __bf16 v8bf;
typedef __attribute__((ext_vector_type(8)))  float  v8f;

#define TWO_PI 6.28318530717958647692f

// ---------------------------------------------------------------------
// Kernel 1: warp-resampling tables (depend only on dlnf): 16 x 2048
// ---------------------------------------------------------------------
__global__ void chirp_tables(const float* __restrict__ dlnf,
                             int* __restrict__ t_lo,
                             float* __restrict__ t_wl,
                             float* __restrict__ t_wh) {
    int e = blockIdx.x * blockDim.x + threadIdx.x;
    if (e >= D_DIM * KTAU) return;
    int d = e >> 11;
    int t = e & (KTAU - 1);

    float beta  = 2.0f * dlnf[d];
    bool  small = fabsf(beta) < 1e-8f;
    float bs    = small ? 1e-8f : beta;
    float e2b   = expf(2.0f * bs);

    float tauf = 2.0f * (float)t / (float)KTAU - 1.0f;
    float ts   = log1pf((tauf + 1.0f) * 0.5f * (e2b - 1.0f)) / bs - 1.0f;
    if (small) ts = tauf;
    float tm   = log1pf(0.5f * (e2b - 1.0f)) / bs - 1.0f;
    if (small) tm = 0.0f;
    float jac  = small ? 1.0f : expf(-bs * (ts - tm));

    float idxf = 0.5f * (float)K_WIN * (ts + 1.0f);
    int lo = (int)floorf(idxf);
    lo = lo < 0 ? 0 : (lo > (K_WIN - 2) ? (K_WIN - 2) : lo);
    float frac = idxf - (float)lo;

    t_lo[e] = lo;
    t_wl[e] = (1.0f - frac) * jac;
    t_wh[e] = frac * jac;
}

// ---------------------------------------------------------------------
// Kernel 2: rDFT basis, pre-swizzled into WMMA B-fragment order,
// split into bf16 hi/lo.  Element (ct, kc, lane, j):
//   K   = kc*32 + (lane>=16 ? 16 : 0) + j      (contiguous k order, same as A)
//   col = ct*16 + (lane & 15);  col=2f -> cos, col=2f+1 -> -sin
// ---------------------------------------------------------------------
__global__ void chirp_basis(__hip_bfloat16* __restrict__ bhi_,
                            __hip_bfloat16* __restrict__ blo_) {
    __bf16* b_hi = reinterpret_cast<__bf16*>(bhi_);
    __bf16* b_lo = reinterpret_cast<__bf16*>(blo_);
    int idx = blockIdx.x * blockDim.x + threadIdx.x;
    if (idx >= NT_CT * NKC * 32 * 16) return;

    int j    = idx & 15;
    int lane = (idx >> 4) & 31;
    int kc   = (idx >> 9) & (NKC - 1);
    int ct   = idx >> 15;

    int K   = kc * 32 + ((lane & 16) ? 16 : 0) + j;
    int col = ct * 16 + (lane & 15);

    float val = 0.0f;
    if (col < NCOL) {
        int f  = col >> 1;
        int ph = (f * K) & (KTAU - 1);            // exact integer angle reduction
        float th = (float)ph * (TWO_PI / (float)KTAU);
        val = (col & 1) ? -sinf(th) : cosf(th);
    }
    __bf16 h = (__bf16)val;
    b_hi[idx] = h;
    b_lo[idx] = (__bf16)(val - (float)h);
}

// ---------------------------------------------------------------------
// Kernel 3: fused window -> resample -> rDFT GEMM (bf16x3 WMMA).
// One workgroup per (b, window); 8 waves sweep 129 column tiles.
// ---------------------------------------------------------------------
__global__ void __launch_bounds__(256, 1)
chirp_main(const float* __restrict__ x,
           const int* __restrict__ t_lo,
           const float* __restrict__ t_wl,
           const float* __restrict__ t_wh,
           const __hip_bfloat16* __restrict__ bhi_,
           const __hip_bfloat16* __restrict__ blo_,
           float* __restrict__ out) {
    const __bf16* b_hi = reinterpret_cast<const __bf16*>(bhi_);
    const __bf16* b_lo = reinterpret_cast<const __bf16*>(blo_);

    extern __shared__ char smem[];
    float*  winbuf = (float*)smem;                                   // 1024 f32
    __bf16* a_hi   = (__bf16*)(smem + 4096);                         // 16 x 2056 bf16
    __bf16* a_lo   = (__bf16*)(smem + 4096 + D_DIM * A_STRIDE * 2);  // 16 x 2056 bf16

    const int tid = threadIdx.x;
    const int bw  = blockIdx.x;          // 0 .. NB*NWIN-1
    const int b   = bw / NWIN;
    const int w   = bw - b * NWIN;
    const float* xs = x + (size_t)b * NSAMP + (size_t)w * HOP;

    // Phase A: Hann-windowed input into LDS
    for (int k = tid; k < K_WIN; k += 256) {
        float wn = 0.5f - 0.5f * __cosf((TWO_PI / (float)K_WIN) * (float)k);
        winbuf[k] = xs[k] * wn;
    }
    __syncthreads();

    // Phase B: resample all 16 warp channels; split into bf16 hi/lo
    for (int e = tid; e < D_DIM * KTAU; e += 256) {
        int d = e >> 11;
        int t = e & (KTAU - 1);
        int lo = t_lo[e];
        float r = t_wl[e] * winbuf[lo] + t_wh[e] * winbuf[lo + 1];
        __bf16 h = (__bf16)r;
        a_hi[d * A_STRIDE + t] = h;
        a_lo[d * A_STRIDE + t] = (__bf16)(r - (float)h);
    }
    __syncthreads();

    // Phase C: 16x16 output tiles, K-loop over 64 chunks of 32, bf16x3
    const int lane   = tid & 31;
    const int wid    = tid >> 5;
    const int arow   = lane & 15;                // A-matrix M row for this lane
    const int khalfA = (lane & 16) ? 16 : 0;     // lanes 16..31 take K 16..31 (contiguous, matches B)
    float* outp = out + (size_t)bw * (D_DIM * NCOL);

    for (int ct = wid; ct < NT_CT; ct += 8) {
        v8f acc = {};
        for (int kc = 0; kc < NKC; ++kc) {
            const __bf16* ah_row = a_hi + arow * A_STRIDE + kc * 32;
            const __bf16* al_row = a_lo + arow * A_STRIDE + kc * 32;
            // elements 0..7 -> K = khalfA+0..7 ; 8..15 -> K = khalfA+8..15
            v8bf h0 = *(const v8bf*)(ah_row + khalfA);
            v8bf h1 = *(const v8bf*)(ah_row + 8 + khalfA);
            v8bf l0 = *(const v8bf*)(al_row + khalfA);
            v8bf l1 = *(const v8bf*)(al_row + 8 + khalfA);
            v16bf ah = __builtin_shufflevector(h0, h1, 0,1,2,3,4,5,6,7,8,9,10,11,12,13,14,15);
            v16bf al = __builtin_shufflevector(l0, l1, 0,1,2,3,4,5,6,7,8,9,10,11,12,13,14,15);

            size_t bidx = (((size_t)ct * NKC + kc) * 32 + lane) * 16;
            v16bf bh = *(const v16bf*)(b_hi + bidx);
            v16bf bl = *(const v16bf*)(b_lo + bidx);
            if (kc + 1 < NKC)
                __builtin_prefetch(b_hi + bidx + 32 * 16, 0, 1);   // global_prefetch_b8

            // bf16x3: Ah*Bh + Al*Bh + Ah*Bl  (f32 accumulate)
            acc = __builtin_amdgcn_wmma_f32_16x16x32_bf16(false, ah, false, bh,
                                                          (short)0, acc, false, false);
            acc = __builtin_amdgcn_wmma_f32_16x16x32_bf16(false, al, false, bh,
                                                          (short)0, acc, false, false);
            acc = __builtin_amdgcn_wmma_f32_16x16x32_bf16(false, ah, false, bl,
                                                          (short)0, acc, false, false);
        }
        // C/D layout: acc[r] -> row M=(lane>=16?8:0)+r, col N=lane&15
        int c = ct * 16 + (lane & 15);
        if (c < NCOL) {
            int mbase = (lane & 16) ? 8 : 0;
#pragma unroll
            for (int r = 0; r < 8; ++r)
                outp[(size_t)(mbase + r) * NCOL + c] = acc[r];
        }
    }
}

// ---------------------------------------------------------------------
// Host entry
// ---------------------------------------------------------------------
extern "C" void kernel_launch(void* const* d_in, const int* in_sizes, int n_in,
                              void* d_out, int out_size, void* d_ws, size_t ws_size,
                              hipStream_t stream) {
    const float* x    = (const float*)d_in[0];   // (4, 262144) f32
    const float* dlnf = (const float*)d_in[1];   // (16,) f32
    float* out = (float*)d_out;                  // (4,511,16,1025) complex64 -> interleaved f32

    // workspace layout (bytes)
    char* ws = (char*)d_ws;
    int*   t_lo = (int*)  (ws);                       // 32768 * 4   = 128 KB
    float* t_wl = (float*)(ws + 131072);              // 128 KB
    float* t_wh = (float*)(ws + 262144);              // 128 KB
    __hip_bfloat16* b_hi = (__hip_bfloat16*)(ws + 393216);             // 8.45 MB
    __hip_bfloat16* b_lo = (__hip_bfloat16*)(ws + 393216 + 8454144);   // 8.45 MB

    chirp_tables<<<(D_DIM * KTAU + 255) / 256, 256, 0, stream>>>(dlnf, t_lo, t_wl, t_wh);

    const int nbasis = NT_CT * NKC * 32 * 16;
    chirp_basis<<<(nbasis + 255) / 256, 256, 0, stream>>>(b_hi, b_lo);

    const size_t lds = 4096 + 2 * (size_t)D_DIM * A_STRIDE * 2;   // 135,680 B (< 320 KB/WGP)
    hipFuncSetAttribute((const void*)chirp_main,
                        hipFuncAttributeMaxDynamicSharedMemorySize, (int)lds);
    chirp_main<<<NB * NWIN, 256, lds, stream>>>(x, t_lo, t_wl, t_wh, b_hi, b_lo, out);
}
